// TextImageMix_38414187495819
// MI455X (gfx1250) — matrix operand order repaired and checked by reading, not compiled, not run on hardware
//
#include <hip/hip_runtime.h>
#include <hip/hip_bf16.h>

typedef __attribute__((ext_vector_type(16))) __bf16 v16bf;
typedef __attribute__((ext_vector_type(8)))  float  v8f;

#define DIM   1024
#define NSAMP 8192
#define KDIM  2048

// ---------------- bf16 helpers (RNE, self-contained) ----------------
__device__ __forceinline__ unsigned short f2bf(float f) {
    unsigned int u = __float_as_uint(f);
    unsigned int r = u + 0x7FFFu + ((u >> 16) & 1u);
    if ((u & 0x7F800000u) == 0x7F800000u) r = u;   // inf/NaN passthrough
    return (unsigned short)(r >> 16);
}

__device__ __forceinline__ float gelu_exact(float x) {
    return 0.5f * x * (1.0f + erff(x * 0.70710678118654752440f));
}

struct Q2 { uint4 a, b; };
__device__ __forceinline__ v16bf mkfrag(uint4 a, uint4 b) {
    Q2 q; q.a = a; q.b = b;
    return __builtin_bit_cast(v16bf, q);
}

// LDS-only workgroup barrier: waits only DScnt (LDS) before signaling, so
// in-flight GLOBAL loads (which land in private VGPRs and are consumed by
// compiler-tracked ds_stores next iteration) stay outstanding across the
// barrier instead of being drained by __syncthreads()' implicit global fence.
__device__ __forceinline__ void lds_barrier() {
    asm volatile("s_wait_dscnt 0\n\t"
                 "s_barrier_signal -1\n\t"
                 "s_barrier_wait -1" ::: "memory");
}

// ---------------- kernel 1: fold conv_b + tfeat rows into base[2][1024] ----
__global__ void prep_base_kernel(const float* __restrict__ tn,
                                 const float* __restrict__ ta,
                                 const float* __restrict__ cw,
                                 const float* __restrict__ cb,
                                 float* __restrict__ baseW) {
    int k = blockIdx.x * 256 + threadIdx.x;
    if (k >= 2 * DIM) return;
    int c = k >> 10, d = k & (DIM - 1);
    float s = cb[c];
#pragma unroll
    for (int h = 0; h < 2; ++h) {
        const float* T = h ? ta : tn;
#pragma unroll
        for (int w = 0; w < 5; ++w) {
            int dd = d + w - 2;
            if (dd >= 0 && dd < DIM) s += cw[c * 15 + h * 5 + w] * T[dd];
        }
    }
    baseW[k] = s;
}

// ---------------- kernel 2: w2 fp32 -> bf16 (K-contiguous) ----------------
__global__ void prep_w2_kernel(const float* __restrict__ w2,
                               unsigned short* __restrict__ w2b) {
    size_t idx = (size_t)blockIdx.x * 256 + threadIdx.x;   // 524288 threads
    size_t o = idx * 4;
    float4 v = *(const float4*)(w2 + o);
    ushort4 u;
    u.x = f2bf(v.x); u.y = f2bf(v.y); u.z = f2bf(v.z); u.w = f2bf(v.w);
    *(ushort4*)(w2b + o) = u;
}

// ---------------- kernel 3: conv tail + GELU -> bf16 activations ----------
__global__ void prep_act_kernel(const float* __restrict__ ifeats,
                                const float* __restrict__ cw,
                                const float* __restrict__ baseW,
                                unsigned short* __restrict__ act) {
    __shared__ float rowp[DIM + 4];        // zero-padded halo of 2 each side
    const int i = blockIdx.x;
    const int t = threadIdx.x;
    const float* row = ifeats + (size_t)i * DIM;
    float4 v = *(const float4*)(row + t * 4);
    rowp[2 + t * 4 + 0] = v.x;
    rowp[2 + t * 4 + 1] = v.y;
    rowp[2 + t * 4 + 2] = v.z;
    rowp[2 + t * 4 + 3] = v.w;
    if (t == 0) { rowp[0] = 0.f; rowp[1] = 0.f; rowp[DIM + 2] = 0.f; rowp[DIM + 3] = 0.f; }
    __syncthreads();

    float c0[5], c1[5];
#pragma unroll
    for (int w = 0; w < 5; ++w) { c0[w] = cw[10 + w]; c1[w] = cw[25 + w]; }

    unsigned short* arow = act + (size_t)i * KDIM;
#pragma unroll
    for (int u = 0; u < 4; ++u) {
        int d = t * 4 + u;
        float s0 = baseW[d];
        float s1 = baseW[DIM + d];
#pragma unroll
        for (int w = 0; w < 5; ++w) {
            float x = rowp[d + w];
            s0 += c0[w] * x;
            s1 += c1[w] * x;
        }
        arow[d]       = f2bf(gelu_exact(s0));
        arow[DIM + d] = f2bf(gelu_exact(s1));
    }
}

// ---------------- kernel 4: bf16 WMMA GEMM + epilogue ---------------------
// out[i][j] = ifeats[i][j] + b2[j] + sum_k act[i][k] * w2[j][k]
// Block tile 256(M) x 128(N), BK=32. 8 waves in 4x2 grid, each wave owns a
// 64x64 macro-tile = 4x4 WMMA tiles -> 16 wmma per K-step vs 16 ds_load_b128.
#define BM 256
#define BN 128
#define BK 32
#define SSTR 40   // padded LDS row stride (elements): 80B -> conflict-free b128

__global__ __launch_bounds__(256)
void gemm_kernel(const unsigned short* __restrict__ A,   // act [8192][2048]
                 const unsigned short* __restrict__ B,   // w2b [1024][2048]
                 const float* __restrict__ ifeats,
                 const float* __restrict__ b2,
                 float* __restrict__ out) {
    __shared__ __align__(16) unsigned short As[2][BM * SSTR];  // 2 x 20480 B
    __shared__ __align__(16) unsigned short Bs[2][BN * SSTR];  // 2 x 10240 B

    const int t    = threadIdx.x;
    const int lane = t & 31;
    const int wave = t >> 5;
    const int wm   = wave >> 1;       // 0..3  (64 rows each)
    const int wn   = wave & 1;        // 0..1  (64 cols each)
    const int m16  = lane & 15;
    const int hi   = lane >> 4;

    const int bm = blockIdx.y * BM;
    const int bn = blockIdx.x * BN;

    // staging: 256 threads cover (256+128) rows x 32 cols of bf16 per K-step
    const int r0 = t >> 2;            // 0..63
    const int c0 = (t & 3) << 3;      // 0,8,16,24

    const unsigned short* aP0 = A + (size_t)(bm + r0)       * KDIM + c0;
    const unsigned short* aP1 = A + (size_t)(bm + r0 + 64)  * KDIM + c0;
    const unsigned short* aP2 = A + (size_t)(bm + r0 + 128) * KDIM + c0;
    const unsigned short* aP3 = A + (size_t)(bm + r0 + 192) * KDIM + c0;
    const unsigned short* bP0 = B + (size_t)(bn + r0)       * KDIM + c0;
    const unsigned short* bP1 = B + (size_t)(bn + r0 + 64)  * KDIM + c0;

    uint4 ra0, ra1, ra2, ra3, rb0, rb1;
    auto gload = [&](int kb) {
        ra0 = *(const uint4*)(aP0 + kb);
        ra1 = *(const uint4*)(aP1 + kb);
        ra2 = *(const uint4*)(aP2 + kb);
        ra3 = *(const uint4*)(aP3 + kb);
        rb0 = *(const uint4*)(bP0 + kb);
        rb1 = *(const uint4*)(bP1 + kb);
    };
    auto sstore = [&](int buf) {
        *(uint4*)&As[buf][(size_t)r0         * SSTR + c0] = ra0;
        *(uint4*)&As[buf][(size_t)(r0 + 64)  * SSTR + c0] = ra1;
        *(uint4*)&As[buf][(size_t)(r0 + 128) * SSTR + c0] = ra2;
        *(uint4*)&As[buf][(size_t)(r0 + 192) * SSTR + c0] = ra3;
        *(uint4*)&Bs[buf][(size_t)r0         * SSTR + c0] = rb0;
        *(uint4*)&Bs[buf][(size_t)(r0 + 64)  * SSTR + c0] = rb1;
    };

    v8f acc[4][4] = {};

    gload(0);
    sstore(0);
    gload(BK);
    lds_barrier();

    const int KT = KDIM / BK;   // 64
    for (int kt = 0; kt < KT; ++kt) {
        const int cur = kt & 1;
        if (kt + 1 < KT) sstore(cur ^ 1);          // stage regs loaded last iter
        if (kt + 2 < KT) gload((kt + 2) * BK);     // pipeline next global loads

        // B fragments first (resident across the tm loop):
        // lane holds B[hi*16 .. +15, n] (K-major, contiguous 32B)
        v16bf fb[4];
#pragma unroll
        for (int tn = 0; tn < 4; ++tn) {
            const unsigned short* p = &Bs[cur][(size_t)(wn * 64 + tn * 16 + m16) * SSTR + hi * 16];
            fb[tn] = mkfrag(*(const uint4*)p, *(const uint4*)(p + 8));
        }
        // A fragments one at a time to cap register pressure:
        // lane holds A[m, hi*16 .. +15]
#pragma unroll
        for (int tm = 0; tm < 4; ++tm) {
            const unsigned short* p = &As[cur][(size_t)(wm * 64 + tm * 16 + m16) * SSTR + hi * 16];
            v16bf fa = mkfrag(*(const uint4*)p, *(const uint4*)(p + 8));
#pragma unroll
            for (int tn = 0; tn < 4; ++tn)
                acc[tm][tn] = __builtin_amdgcn_wmma_f32_16x16x32_bf16(
                    false, fa, false, fb[tn], (short)0, acc[tm][tn], false, false);
        }

        lds_barrier();
    }

    // epilogue: D[m,n] -> VGPR r: (M = r + 8*hi, N = lane&15)
#pragma unroll
    for (int tm = 0; tm < 4; ++tm) {
        const int ibase = bm + wm * 64 + tm * 16 + 8 * hi;
#pragma unroll
        for (int tn = 0; tn < 4; ++tn) {
            const int j = bn + wn * 64 + tn * 16 + m16;
            const float bj = b2[j];
#pragma unroll
            for (int r = 0; r < 8; ++r) {
                const size_t o = (size_t)(ibase + r) * DIM + j;
                out[o] = acc[tm][tn][r] + bj + ifeats[o];
            }
        }
    }
}

// ---------------- launcher ------------------------------------------------
extern "C" void kernel_launch(void* const* d_in, const int* in_sizes, int n_in,
                              void* d_out, int out_size, void* d_ws, size_t ws_size,
                              hipStream_t stream) {
    const float* ifeats  = (const float*)d_in[0];
    const float* tfeat_n = (const float*)d_in[1];
    const float* tfeat_a = (const float*)d_in[2];
    const float* conv_w  = (const float*)d_in[3];
    const float* conv_b  = (const float*)d_in[4];
    const float* w2      = (const float*)d_in[5];
    const float* b2      = (const float*)d_in[6];
    float* out = (float*)d_out;

    // workspace layout
    unsigned short* w2b  = (unsigned short*)d_ws;                    // 4 MB
    unsigned short* act  = w2b + (size_t)DIM * KDIM;                 // 32 MB
    float*          base = (float*)(act + (size_t)NSAMP * KDIM);    // 8 KB

    prep_base_kernel<<<8, 256, 0, stream>>>(tfeat_n, tfeat_a, conv_w, conv_b, base);
    prep_w2_kernel<<<(DIM * KDIM) / (256 * 4), 256, 0, stream>>>(w2, w2b);
    prep_act_kernel<<<NSAMP, 256, 0, stream>>>(ifeats, conv_w, base, act);
    gemm_kernel<<<dim3(DIM / BN, NSAMP / BM), 256, 0, stream>>>(act, w2b, ifeats, b2, out);
}
